// Attention_54039278518650
// MI455X (gfx1250) — matrix operand order repaired and checked by reading, not compiled, not run on hardware
//
#include <hip/hip_runtime.h>

// ---------- types ----------
typedef __attribute__((ext_vector_type(16))) _Float16 v16h;
typedef __attribute__((ext_vector_type(8)))  float    v8f;
typedef __attribute__((ext_vector_type(4)))  unsigned int u32x4;
typedef __attribute__((ext_vector_type(8)))  unsigned int u32x8;

union AF { v16h v; u32x4 u[2]; };
union BF { v16h v; u32x4 u[2]; };

#define WMMA(a, b, c) \
  __builtin_amdgcn_wmma_f32_16x16x32_f16(false, (a), false, (b), (short)0, (c), false, false)

__device__ __forceinline__ v8f v8f_zero() {
  v8f z;
#pragma unroll
  for (int i = 0; i < 8; ++i) z[i] = 0.0f;
  return z;
}

// LDS byte offset of a __shared__ pointer (low 32 bits of the flat address)
__device__ __forceinline__ unsigned lds_off(const void* p) {
  return (unsigned)(unsigned long long)p;
}

// ---------- problem constants ----------
// B=2, L=1024, P=1024, T=2048, HID=2048, NH=32, NKV=8, D=64, N_REPEAT=4
#define C_HID  2048
#define C_NQK  3072           // NH*D + 2*NKV*D

// ---------------------------------------------------------------
// fp32 -> f16 cast with row restriding (x, W packs, past_k/v -> cache)
// ---------------------------------------------------------------
__global__ void pack_cast(const float* __restrict__ src, _Float16* __restrict__ dst,
                          int n, int cols, int ld, int off) {
  for (int i = blockIdx.x * blockDim.x + threadIdx.x; i < n; i += gridDim.x * blockDim.x) {
    int r = i / cols, c = i - r * cols;
    dst[(size_t)r * ld + off + c] = (_Float16)src[i];
  }
}

// ---------------------------------------------------------------
// Shared WMMA GEMM mainloop: C[128x128] += A[128xK] * B[KxN] tile
// block = 256 threads = 8 waves (4 m-waves x 2 n-waves), wave tile 32x64
// A tile staged with GLOBAL_LOAD_ASYNC_TO_LDS_B128 (ASYNCcnt path).
// ---------------------------------------------------------------
__device__ __forceinline__ void gemm_tiles(
    const _Float16* __restrict__ A, int lda,
    const _Float16* __restrict__ B, int ldb,
    int K, int blockM, int blockN,
    _Float16 (*As)[32], _Float16 (*Bs)[32],
    v8f acc[2][4]) {
  const int tid  = threadIdx.x;
  const int lane = tid & 31;
  const int hi   = lane >> 4;
  const int ln   = lane & 15;
  const int wave = tid >> 5;
  const int mw   = wave & 3;
  const int nw   = wave >> 2;

  for (int k0 = 0; k0 < K; k0 += 32) {
    // stage A tile: 128x32 halves; async global->LDS, 16B per thread-chunk
#pragma unroll
    for (int c = 0; c < 2; ++c) {
      int idx = tid + c * 256;          // 0..511
      int r   = idx >> 2;               // 0..127
      int cc  = (idx & 3) << 3;         // 0,8,16,24
      unsigned goff  = (unsigned)(((blockM + r) * lda + k0 + cc) * 2);
      unsigned laddr = lds_off(&As[r][cc]);
      asm volatile("global_load_async_to_lds_b128 %0, %1, %2"
                   :: "v"(laddr), "v"(goff), "s"(A) : "memory");
    }
    // stage B tile transposed: Bs[n][k] = B[k0+k][blockN+n]
#pragma unroll
    for (int c = 0; c < 2; ++c) {
      int idx = tid + c * 256;          // 0..511
      int kk  = idx >> 4;               // 0..31
      int nn  = (idx & 15) << 3;        // 0..120
      u32x4 dv = *(const u32x4*)(B + (size_t)(k0 + kk) * ldb + blockN + nn);
      union { u32x4 u; _Float16 h[8]; } t; t.u = dv;
#pragma unroll
      for (int j = 0; j < 8; ++j) Bs[nn + j][kk] = t.h[j];
    }
    if (k0 + 32 < K) {
      __builtin_prefetch(B + (size_t)(k0 + 32 + (tid >> 4)) * ldb + blockN, 0, 0);
    }
    asm volatile("s_wait_asynccnt 0x0" ::: "memory");
    __syncthreads();

    AF a[2];
#pragma unroll
    for (int mi = 0; mi < 2; ++mi) {
      int row = mw * 32 + mi * 16 + ln;               // A lane row = lane&15
      a[mi].u[0] = *(const u32x4*)(&As[row][hi * 8]);       // K = hi*8 .. +7
      a[mi].u[1] = *(const u32x4*)(&As[row][16 + hi * 8]);  // K = 16+hi*8 ..
    }
    BF bf[4];
#pragma unroll
    for (int ni = 0; ni < 4; ++ni) {
      int coln = nw * 64 + ni * 16 + ln;
      bf[ni].u[0] = *(const u32x4*)(&Bs[coln][hi * 8]);       // K = hi*8 .. +7
      bf[ni].u[1] = *(const u32x4*)(&Bs[coln][16 + hi * 8]);  // K = 16+hi*8 ..
    }
#pragma unroll
    for (int mi = 0; mi < 2; ++mi)
#pragma unroll
      for (int ni = 0; ni < 4; ++ni)
        acc[mi][ni] = WMMA(a[mi].v, bf[ni].v, acc[mi][ni]);
    __syncthreads();
  }
}

// ---------------------------------------------------------------
// Fused QKV GEMM + bias + RoPE + scatter to q / k-cache / v-cache
// A = x_f16 [2048][2048], B = [Wq|Wk|Wv]_f16 [2048][3072]
// ---------------------------------------------------------------
__global__ __launch_bounds__(256) void gemm_qkv(
    const _Float16* __restrict__ Xh, const _Float16* __restrict__ Wh,
    const float* __restrict__ bq, const float* __restrict__ bk,
    const float* __restrict__ bv,
    const float* __restrict__ cosb, const float* __restrict__ sinb, // [1024][64]
    _Float16* __restrict__ Qh,   // [B][32][1024][64]
    _Float16* __restrict__ Kc,   // [B][8][2048][64]
    _Float16* __restrict__ Vc) { // [B][8][2048][64]
  __shared__ _Float16 As[128][32];
  __shared__ _Float16 Bs[128][32];
  v8f acc[2][4];
#pragma unroll
  for (int mi = 0; mi < 2; ++mi)
#pragma unroll
    for (int ni = 0; ni < 4; ++ni) acc[mi][ni] = v8f_zero();

  const int blockM = blockIdx.y * 128;
  const int blockN = blockIdx.x * 128;
  gemm_tiles(Xh, C_HID, Wh, C_NQK, C_HID, blockM, blockN, As, Bs, acc);

  const int lane = threadIdx.x & 31;
  const int hi = lane >> 4, ln = lane & 15;
  const int wave = threadIdx.x >> 5;
  const int mw = wave & 3, nw = wave >> 2;
  const int colBase = blockN + nw * 64;   // 64-aligned -> one head per wave col

#pragma unroll
  for (int mi = 0; mi < 2; ++mi) {
#pragma unroll
    for (int ni = 0; ni < 4; ++ni) {
#pragma unroll
      for (int r = 0; r < 8; ++r) {
        int row = blockM + mw * 32 + mi * 16 + hi * 8 + r;
        int col = colBase + ni * 16 + ln;
        int bb = row >> 10;          // batch
        int l  = row & 1023;         // seq pos
        float v  = acc[mi][ni][r];
        float vp = acc[mi][ni ^ 2][r];   // partner element d^32 (same head)
        if (col < 2048) {                 // ---- Q + RoPE ----
          int d = col & 63, h = col >> 6;
          v  += bq[col];
          vp += bq[col ^ 32];
          float cv = cosb[l * 64 + d], sv = sinb[l * 64 + d];
          float o = v * cv + ((d < 32) ? -vp : vp) * sv;
          Qh[(((size_t)bb * 32 + h) * 1024 + l) * 64 + d] = (_Float16)o;
        } else if (col < 2560) {          // ---- K + RoPE -> cache tail ----
          int c2 = col - 2048;
          int d = c2 & 63, hk = c2 >> 6;
          v  += bk[c2];
          vp += bk[c2 ^ 32];
          float cv = cosb[l * 64 + d], sv = sinb[l * 64 + d];
          float o = v * cv + ((d < 32) ? -vp : vp) * sv;
          Kc[(((size_t)bb * 8 + hk) * 2048 + 1024 + l) * 64 + d] = (_Float16)o;
        } else {                          // ---- V -> cache tail ----
          int c2 = col - 2560;
          int d = c2 & 63, hk = c2 >> 6;
          v += bv[c2];
          Vc[(((size_t)bb * 8 + hk) * 2048 + 1024 + l) * 64 + d] = (_Float16)v;
        }
      }
    }
  }
}

// ---------------------------------------------------------------
// Flash attention: one (b, h, 128-row q tile) per block; GQA h->h/4.
// K chunk staged by the Tensor Data Mover (TENSORcnt path); V staged
// transposed by VALU; online softmax; WMMA for S=QK^T and O=P*V.
// ---------------------------------------------------------------
__global__ __launch_bounds__(256) void attn_kernel(
    const _Float16* __restrict__ Qh, const _Float16* __restrict__ Kc,
    const _Float16* __restrict__ Vc, _Float16* __restrict__ Oh) {
  __shared__ _Float16 Ks[32][64];      // [t][d]  (TDM destination)
  __shared__ _Float16 Vs[64][32];      // [d][t]  (transposed)
  __shared__ _Float16 Ps[8][16][32];   // per-wave P transpose scratch

  const int tid = threadIdx.x;
  const int lane = tid & 31;
  const int hi = lane >> 4, ln = lane & 15;
  const int wave = tid >> 5;
  const int l0 = blockIdx.x * 128;
  const int h = blockIdx.y, b = blockIdx.z;
  const int hkv = h >> 2;

  // Q A-fragments for this wave's 16 rows (K-dim = d = 64, two 32-steps)
  const _Float16* qbase =
      Qh + (((size_t)b * 32 + h) * 1024 + l0 + wave * 16) * 64;
  AF aq[2];
#pragma unroll
  for (int kk = 0; kk < 2; ++kk) {
    aq[kk].u[0] = *(const u32x4*)(qbase + (size_t)ln * 64 + kk * 32 + hi * 8);
    aq[kk].u[1] = *(const u32x4*)(qbase + (size_t)ln * 64 + kk * 32 + 16 + hi * 8);
  }

  const _Float16* kbase = Kc + ((size_t)b * 8 + hkv) * 2048 * 64;
  const _Float16* vbase = Vc + ((size_t)b * 8 + hkv) * 2048 * 64;
  const unsigned ks_lds = lds_off(&Ks[0][0]);

  float mrow[8], lrow[8];
  v8f o[4];
#pragma unroll
  for (int r = 0; r < 8; ++r) { mrow[r] = -1e30f; lrow[r] = 0.0f; }
#pragma unroll
  for (int ni = 0; ni < 4; ++ni) o[ni] = v8f_zero();

  const int nch = (1024 + l0 + 128) >> 5;   // causal bound, uniform per block
  for (int ch = 0; ch < nch; ++ch) {
    const int t0 = ch << 5;

    // ---- K chunk [32][64] via Tensor Data Mover (wave 0 issues) ----
    if (wave == 0) {
      unsigned long long ga =
          (unsigned long long)(const void*)(kbase + (size_t)t0 * 64);
      u32x4 g0;
      g0[0] = 1u;                                   // count=1, user descriptor
      g0[1] = ks_lds;                               // lds_addr (bytes)
      g0[2] = (unsigned)(ga & 0xffffffffu);         // global_addr[31:0]
      g0[3] = (unsigned)((ga >> 32) & 0x01ffffffu)  // global_addr[56:32]
              | 0x80000000u;                        // type=2 ("image")
      u32x8 g1;
      g1[0] = 1u << 16;           // data_size=1 -> 2-byte elements
      g1[1] = 64u << 16;          // tensor_dim0 = 64   (bits 79:48, lo16)
      g1[2] = 2048u << 16;        // dim0 hi16=0 ; tensor_dim1 = 2048 (lo16)
      g1[3] = 64u << 16;          // dim1 hi16=0 ; tile_dim0 = 64
      g1[4] = 32u;                // tile_dim1 = 32 ; tile_dim2 = 0
      g1[5] = 64u;                // tensor_dim0_stride = 64 (lo32)
      g1[6] = 0u;                 // stride hi16 ; dim1_stride lo16
      g1[7] = 0u;                 // dim1_stride hi32
      asm volatile("tensor_load_to_lds %0, %1" :: "s"(g0), "s"(g1) : "memory");
    }

    // ---- V chunk transposed [64][32] (all threads) ----
    {
      int t = tid >> 3;
      int c = (tid & 7) << 3;
      u32x4 dv = *(const u32x4*)(vbase + (size_t)(t0 + t) * 64 + c);
      union { u32x4 u; _Float16 hh[8]; } tv; tv.u = dv;
#pragma unroll
      for (int j = 0; j < 8; ++j) Vs[c + j][t] = tv.hh[j];
    }
    if (wave == 0) __builtin_amdgcn_s_wait_tensorcnt(0);
    __syncthreads();

    // S = Q * K^T (two 16-wide t tiles, K-dim 64 = 2 wmma each)
    v8f s[2];
#pragma unroll
    for (int ni = 0; ni < 2; ++ni) {
      int t = ni * 16 + ln;
      BF b0, b1;
      b0.u[0] = *(const u32x4*)(&Ks[t][hi * 8]);
      b0.u[1] = *(const u32x4*)(&Ks[t][16 + hi * 8]);
      b1.u[0] = *(const u32x4*)(&Ks[t][32 + hi * 8]);
      b1.u[1] = *(const u32x4*)(&Ks[t][48 + hi * 8]);
      v8f z = v8f_zero();
      s[ni] = WMMA(aq[0].v, b0.v, z);
      s[ni] = WMMA(aq[1].v, b1.v, s[ni]);
    }

    // scale + causal mask + online softmax update; stash P in LDS
#pragma unroll
    for (int r = 0; r < 8; ++r) {
      int lr = l0 + wave * 16 + hi * 8 + r;     // query row
      float s0 = s[0][r] * 0.125f;
      float s1 = s[1][r] * 0.125f;
      s0 = ((t0 + ln) > 1024 + lr) ? -1e9f : s0;
      s1 = ((t0 + 16 + ln) > 1024 + lr) ? -1e9f : s1;
      float mx = fmaxf(s0, s1);
#pragma unroll
      for (int msk = 1; msk <= 8; msk <<= 1)
        mx = fmaxf(mx, __shfl_xor(mx, msk, 32));
      float mnew = fmaxf(mrow[r], mx);
      float corr = __expf(mrow[r] - mnew);
      float p0 = __expf(s0 - mnew);
      float p1 = __expf(s1 - mnew);
      float ps = p0 + p1;
#pragma unroll
      for (int msk = 1; msk <= 8; msk <<= 1)
        ps += __shfl_xor(ps, msk, 32);
      lrow[r] = lrow[r] * corr + ps;
      mrow[r] = mnew;
#pragma unroll
      for (int ni = 0; ni < 4; ++ni) o[ni][r] *= corr;
      Ps[wave][hi * 8 + r][ln]      = (_Float16)p0;
      Ps[wave][hi * 8 + r][16 + ln] = (_Float16)p1;
    }

    // P as A-fragment (same-wave LDS round-trip transposes C->A layout)
    AF ap;
    ap.u[0] = *(const u32x4*)(&Ps[wave][ln][hi * 8]);
    ap.u[1] = *(const u32x4*)(&Ps[wave][ln][16 + hi * 8]);

    // O += P * V  (K-dim = 32, four 16-wide d tiles)
#pragma unroll
    for (int ni = 0; ni < 4; ++ni) {
      BF bv_;
      bv_.u[0] = *(const u32x4*)(&Vs[ni * 16 + ln][hi * 8]);
      bv_.u[1] = *(const u32x4*)(&Vs[ni * 16 + ln][16 + hi * 8]);
      o[ni] = WMMA(ap.v, bv_.v, o[ni]);
    }
    __syncthreads();
  }

  // normalize + write attn output as f16 row-major [B*L][2048]
#pragma unroll
  for (int ni = 0; ni < 4; ++ni)
#pragma unroll
    for (int r = 0; r < 8; ++r) {
      int lr = l0 + wave * 16 + hi * 8 + r;
      float v = o[ni][r] / lrow[r];
      Oh[((size_t)b * 1024 + lr) * 2048 + h * 64 + ni * 16 + ln] = (_Float16)v;
    }
}

// ---------------------------------------------------------------
// Output projection: attn_f16 [2048][2048] @ Wo_f16 [2048][2048] -> fp32
// ---------------------------------------------------------------
__global__ __launch_bounds__(256) void gemm_out_k(
    const _Float16* __restrict__ Ah, const _Float16* __restrict__ Bh,
    float* __restrict__ out) {
  __shared__ _Float16 As[128][32];
  __shared__ _Float16 Bs[128][32];
  v8f acc[2][4];
#pragma unroll
  for (int mi = 0; mi < 2; ++mi)
#pragma unroll
    for (int ni = 0; ni < 4; ++ni) acc[mi][ni] = v8f_zero();

  const int blockM = blockIdx.y * 128;
  const int blockN = blockIdx.x * 128;
  gemm_tiles(Ah, 2048, Bh, 2048, 2048, blockM, blockN, As, Bs, acc);

  const int lane = threadIdx.x & 31;
  const int hi = lane >> 4, ln = lane & 15;
  const int wave = threadIdx.x >> 5;
  const int mw = wave & 3, nw = wave >> 2;
#pragma unroll
  for (int mi = 0; mi < 2; ++mi)
#pragma unroll
    for (int ni = 0; ni < 4; ++ni)
#pragma unroll
      for (int r = 0; r < 8; ++r) {
        int row = blockM + mw * 32 + mi * 16 + hi * 8 + r;
        int col = blockN + nw * 64 + ni * 16 + ln;
        out[(size_t)row * 2048 + col] = acc[mi][ni][r];
      }
}

// ---------------------------------------------------------------
extern "C" void kernel_launch(void* const* d_in, const int* in_sizes, int n_in,
                              void* d_out, int out_size, void* d_ws, size_t ws_size,
                              hipStream_t stream) {
  (void)in_sizes; (void)n_in; (void)out_size; (void)ws_size;
  const float* x      = (const float*)d_in[0];
  // d_in[1] attention_mask: pure causal, computed analytically
  const float* cosb   = (const float*)d_in[2];
  const float* sinb   = (const float*)d_in[3];
  const float* past_k = (const float*)d_in[4];
  const float* past_v = (const float*)d_in[5];
  const float* Wq     = (const float*)d_in[6];
  const float* bq     = (const float*)d_in[7];
  const float* Wk     = (const float*)d_in[8];
  const float* bk     = (const float*)d_in[9];
  const float* Wv     = (const float*)d_in[10];
  const float* bv     = (const float*)d_in[11];
  const float* Wo     = (const float*)d_in[12];
  float* out = (float*)d_out;

  char* ws = (char*)d_ws;
  _Float16* Xh    = (_Float16*)(ws + 0);          //  8 MB  [2048][2048]
  _Float16* Wqkvh = (_Float16*)(ws + (8u << 20)); // 12 MB  [2048][3072]
  _Float16* Woh   = (_Float16*)(ws + (20u << 20));//  8 MB  [2048][2048]
  _Float16* Qh    = (_Float16*)(ws + (28u << 20));//  8 MB  [2][32][1024][64]
  _Float16* Kc    = (_Float16*)(ws + (36u << 20));//  4 MB  [2][8][2048][64]
  _Float16* Vc    = (_Float16*)(ws + (40u << 20));//  4 MB
  _Float16* Oh    = (_Float16*)(ws + (44u << 20));//  8 MB  [2048][2048]

  // fp32 -> f16 conversions / packs
  pack_cast<<<4096, 256, 0, stream>>>(x,  Xh,    2048 * 2048, 2048, 2048, 0);
  pack_cast<<<4096, 256, 0, stream>>>(Wq, Wqkvh, 2048 * 2048, 2048, 3072, 0);
  pack_cast<<<2048, 256, 0, stream>>>(Wk, Wqkvh, 2048 * 512,  512,  3072, 2048);
  pack_cast<<<2048, 256, 0, stream>>>(Wv, Wqkvh, 2048 * 512,  512,  3072, 2560);
  pack_cast<<<4096, 256, 0, stream>>>(Wo, Woh,   2048 * 2048, 2048, 2048, 0);
  // past caches -> front half of K/V cache (rows of 65536 -> stride 131072)
  pack_cast<<<2048, 256, 0, stream>>>(past_k, Kc, 16 * 1024 * 64, 65536, 131072, 0);
  pack_cast<<<2048, 256, 0, stream>>>(past_v, Vc, 16 * 1024 * 64, 65536, 131072, 0);

  // fused QKV projection + RoPE + cache scatter
  gemm_qkv<<<dim3(24, 16), 256, 0, stream>>>(Xh, Wqkvh, bq, bk, bv,
                                             cosb, sinb, Qh, Kc, Vc);
  // flash attention
  attn_kernel<<<dim3(8, 32, 2), 256, 0, stream>>>(Qh, Kc, Vc, Oh);
  // output projection -> fp32
  gemm_out_k<<<dim3(16, 16), 256, 0, stream>>>(Oh, Woh, out);
}
